// Net_33243046871554
// MI455X (gfx1250) — matrix operand order repaired and checked by reading, not compiled, not run on hardware
//
#include <hip/hip_runtime.h>
#include <hip/hip_bf16.h>
#include <stdint.h>

// ---------------- CDNA5 WMMA types ----------------
typedef __attribute__((ext_vector_type(16))) __bf16 v16bf;
typedef __attribute__((ext_vector_type(8)))  float  v8f;

__device__ __forceinline__ unsigned short f32_to_bf16_rne(float f) {
    unsigned int u = __float_as_uint(f);
    unsigned int r = u + 0x7FFFu + ((u >> 16) & 1u);   // round-to-nearest-even
    return (unsigned short)(r >> 16);
}
__device__ __forceinline__ unsigned int pack2_bf16(float lo, float hi) {
    return (unsigned int)f32_to_bf16_rne(lo) | ((unsigned int)f32_to_bf16_rne(hi) << 16);
}

// ---------------- GEMM: H = X(MxK) @ W(Kx128), bf16 WMMA, f32 accum ----------------
// 256 threads = 8 waves; block tile 128(M) x 128(N); K staged in 32-chunks.
// Double-buffered LDS + register staging so global loads overlap WMMA compute.
// Both A and B LDS layouts are fragment-contiguous: each lane's 8-dword WMMA
// operand = two 16B-aligned ds_load_b128, conflict-free (stride 20 dwords).
#define MT 128
#define KT 32
#define AS 20   // u32 per A row   (16 kp + 4 pad) -> 80B rows, 16B aligned
#define BS 20   // u32 per B column(16 kp + 4 pad) -> 80B cols, 16B aligned

__device__ __forceinline__ void stage_load(
    const float* __restrict__ X, const float* __restrict__ W,
    int M, int K, int m0, int k0, int tid,
    unsigned int aReg[8], unsigned int bReg[8])
{
#pragma unroll
    for (int i = 0; i < 8; ++i) {                 // A: 128 rows x 16 kp
        int idx  = tid + i * 256;                 // 0..2047
        int r    = idx >> 4;
        int kp   = idx & 15;
        int grow = m0 + r;
        int gk   = k0 + 2 * kp;
        float a0 = 0.f, a1 = 0.f;
        if (grow < M && gk < K) {
            a0 = X[(size_t)grow * K + gk];
            if (gk + 1 < K) a1 = X[(size_t)grow * K + gk + 1];
        }
        aReg[i] = pack2_bf16(a0, a1);
    }
#pragma unroll
    for (int i = 0; i < 8; ++i) {                 // B: 16 kp x 128 n
        int idx = tid + i * 256;
        int kp  = idx >> 7;
        int n   = idx & 127;
        int gk  = k0 + 2 * kp;
        float b0 = 0.f, b1 = 0.f;
        if (gk < K)     b0 = W[(size_t)gk * 128 + n];
        if (gk + 1 < K) b1 = W[(size_t)(gk + 1) * 128 + n];
        bReg[i] = pack2_bf16(b0, b1);
    }
}

__global__ __launch_bounds__(256) void gemm_bf16_wmma(
    const float* __restrict__ X, const float* __restrict__ W,
    float* __restrict__ H, int M, int K)
{
    __shared__ unsigned int Alds[2][MT * AS];   // [row][kp]
    __shared__ unsigned int Blds[2][128 * BS];  // [n][kp]   fragment-contiguous

    const int tid  = threadIdx.x;
    const int wave = tid >> 5;
    const int lane = tid & 31;
    const int m0   = blockIdx.x * MT;

    v8f acc[8];
#pragma unroll
    for (int t = 0; t < 8; ++t) acc[t] = (v8f){0.f,0.f,0.f,0.f,0.f,0.f,0.f,0.f};

    unsigned int aReg[8], bReg[8];
    stage_load(X, W, M, K, m0, 0, tid, aReg, bReg);

    const int nk = (K + KT - 1) / KT;
    for (int kt = 0; kt < nk; ++kt) {
        const int buf = kt & 1;

        // ---- commit staged registers to this K-step's LDS buffer ----
#pragma unroll
        for (int i = 0; i < 8; ++i) {
            int idx = tid + i * 256;
            Alds[buf][(idx >> 4) * AS + (idx & 15)] = aReg[i];
        }
#pragma unroll
        for (int i = 0; i < 8; ++i) {
            int idx = tid + i * 256;
            Blds[buf][(idx & 127) * BS + (idx >> 7)] = bReg[i];
        }
        __syncthreads();

        // ---- prefetch next K-tile into registers (overlaps with WMMA) ----
        if (kt + 1 < nk)
            stage_load(X, W, M, K, m0, (kt + 1) * KT, tid, aReg, bReg);

        // ---- A fragment: lanes 0-15 K {0..7,16..23}, lanes 16-31 {8..15,24..31} ----
        union Frag { v16bf v; unsigned int u[8]; uint4 q[2]; };
        const int mrow = wave * 16 + (lane & 15);
        const int kbpA = (lane < 16) ? 0 : 4;        // pair offset
        Frag afr;
        afr.q[0] = *(const uint4*)&Alds[buf][mrow * AS + kbpA];
        afr.q[1] = *(const uint4*)&Alds[buf][mrow * AS + 8 + kbpA];

        // ---- 8 N-tiles: B fragment lanes 0-15 K {0..7,16..23}, lanes 16-31 {8..15,24..31} ----
        const int ncol = lane & 15;
        const int kbpB = (lane < 16) ? 0 : 4;
#pragma unroll
        for (int t = 0; t < 8; ++t) {
            Frag bfr;
            const unsigned int* bp = &Blds[buf][(t * 16 + ncol) * BS + kbpB];
            bfr.q[0] = *(const uint4*)&bp[0];
            bfr.q[1] = *(const uint4*)&bp[8];
            acc[t] = __builtin_amdgcn_wmma_f32_16x16x32_bf16(
                false, afr.v, false, bfr.v, (short)0, acc[t], false, false);
        }
    }

    // ---- store: C/D layout: VGPR r -> M = r (+8 for lanes 16-31), N = lane%16 ----
    const int rhi  = (lane >> 4) * 8;
    const int ncol = lane & 15;
#pragma unroll
    for (int t = 0; t < 8; ++t) {
#pragma unroll
        for (int r = 0; r < 8; ++r) {
            int grow = m0 + wave * 16 + rhi + r;
            if (grow < M) H[(size_t)grow * 128 + t * 16 + ncol] = acc[t][r];
        }
    }
}

// ---------------- degree / normalization ----------------
__global__ void deg_init(float* deg, int Nn) {
    int i = blockIdx.x * blockDim.x + threadIdx.x;
    if (i < Nn) deg[i] = 1.0f;                 // self-loop weight
}
__global__ void deg_accum(float* deg, const int* __restrict__ dst,
                          const float* __restrict__ w, int E) {
    int e = blockIdx.x * blockDim.x + threadIdx.x;
    if (e < E) atomicAdd(&deg[dst[e]], w[e]);
}
__global__ void deg_fin(float* deg, int Nn) {
    int i = blockIdx.x * blockDim.x + threadIdx.x;
    if (i < Nn) { float d = deg[i]; deg[i] = (d > 0.f) ? rsqrtf(d) : 0.f; }
}
__global__ void norm_k(float* __restrict__ nrm, const float* __restrict__ dinv,
                       const int* __restrict__ src, const int* __restrict__ dst,
                       const float* __restrict__ w, int E) {
    int e = blockIdx.x * blockDim.x + threadIdx.x;
    if (e < E) nrm[e] = w[e] * dinv[src[e]] * dinv[dst[e]];
}

// ---------------- aggregation ----------------
// out = dinv[i]^2 * h[i] + bias   (self loop + bias; serves as init)
__global__ void init_out(float* __restrict__ out, const float* __restrict__ h,
                         const float* __restrict__ dinv, const float* __restrict__ bias,
                         int Nn) {
    int idx = blockIdx.x * blockDim.x + threadIdx.x;     // Nn * 32 float4 lanes
    if (idx >= Nn * 32) return;
    int node = idx >> 5, q = idx & 31;
    float di = dinv[node]; float sl = di * di;
    float4 v = ((const float4*)(h + (size_t)node * 128))[q];
    float4 b = ((const float4*)bias)[q];
    float4 o = { sl * v.x + b.x, sl * v.y + b.y, sl * v.z + b.z, sl * v.w + b.w };
    ((float4*)(out + (size_t)node * 128))[q] = o;
}

// wave per edge: out[dst] += norm[e] * h[src]   (128 feats = 32 lanes x float4)
__global__ __launch_bounds__(256) void edge_scatter(
    float* __restrict__ out, const float* __restrict__ h,
    const float* __restrict__ nrm, const int* __restrict__ src,
    const int* __restrict__ dst, int E)
{
    int wid  = (int)((blockIdx.x * (size_t)blockDim.x + threadIdx.x) >> 5);
    int lane = threadIdx.x & 31;
    if (wid >= E) return;
    int s = src[wid], d = dst[wid];
    float nw = nrm[wid];
    float4 v = ((const float4*)(h + (size_t)s * 128))[lane];
    float* op = out + (size_t)d * 128 + lane * 4;
    atomicAdd(op + 0, nw * v.x);
    atomicAdd(op + 1, nw * v.y);
    atomicAdd(op + 2, nw * v.z);
    atomicAdd(op + 3, nw * v.w);
}

__global__ void relu_k(float* x, int n4) {
    int i = blockIdx.x * blockDim.x + threadIdx.x;
    if (i >= n4) return;
    float4 v = ((float4*)x)[i];
    v.x = fmaxf(v.x, 0.f); v.y = fmaxf(v.y, 0.f);
    v.z = fmaxf(v.z, 0.f); v.w = fmaxf(v.w, 0.f);
    ((float4*)x)[i] = v;
}

// wave per node, in-place log_softmax over 128 feats
__global__ __launch_bounds__(256) void log_softmax_k(float* __restrict__ out, int Nn) {
    int wid  = (int)((blockIdx.x * (size_t)blockDim.x + threadIdx.x) >> 5);
    int lane = threadIdx.x & 31;
    if (wid >= Nn) return;
    float4 v = ((const float4*)(out + (size_t)wid * 128))[lane];
    float m = fmaxf(fmaxf(v.x, v.y), fmaxf(v.z, v.w));
#pragma unroll
    for (int off = 16; off > 0; off >>= 1) m = fmaxf(m, __shfl_xor(m, off, 32));
    float s = __expf(v.x - m) + __expf(v.y - m) + __expf(v.z - m) + __expf(v.w - m);
#pragma unroll
    for (int off = 16; off > 0; off >>= 1) s += __shfl_xor(s, off, 32);
    float ls = m + __logf(s);
    v.x -= ls; v.y -= ls; v.z -= ls; v.w -= ls;
    ((float4*)(out + (size_t)wid * 128))[lane] = v;
}

// ---------------- host orchestration ----------------
static inline size_t align_up(size_t v, size_t a) { return (v + a - 1) & ~(a - 1); }

extern "C" void kernel_launch(void* const* d_in, const int* in_sizes, int n_in,
                              void* d_out, int out_size, void* d_ws, size_t ws_size,
                              hipStream_t stream)
{
    const float* x    = (const float*)d_in[0];
    const int*   ei   = (const int*)d_in[1];
    const float* ea   = (const float*)d_in[2];
    const float* W1   = (const float*)d_in[3];
    const float* b1   = (const float*)d_in[4];
    const float* W2   = (const float*)d_in[5];
    const float* b2   = (const float*)d_in[6];
    const float* W3   = (const float*)d_in[7];
    const float* b3   = (const float*)d_in[8];
    const float* W4   = (const float*)d_in[9];
    const float* b4   = (const float*)d_in[10];
    float* out        = (float*)d_out;

    const int DH  = in_sizes[4];               // 128
    const int DIN = in_sizes[3] / DH;          // 2050
    const int Nn  = in_sizes[0] / DIN;         // 50000
    const int E   = in_sizes[1] / 2;           // 600000
    const int* src = ei;
    const int* dst = ei + E;

    // workspace layout
    char* ws = (char*)d_ws;
    size_t off = 0;
    float* dinv = (float*)(ws + off); off = align_up(off + (size_t)Nn * 4, 256);
    float* nrm  = (float*)(ws + off); off = align_up(off + (size_t)E * 4, 256);
    float* hbuf = (float*)(ws + off); off = align_up(off + (size_t)Nn * 128 * 4, 256);
    float* act  = (float*)(ws + off); off = align_up(off + (size_t)Nn * 128 * 4, 256);
    (void)ws_size; (void)n_in; (void)out_size;

    const int T = 256;
    const int gN   = (Nn + T - 1) / T;
    const int gE   = (E + T - 1) / T;
    const int gEw  = (int)(((size_t)E * 32 + T - 1) / T);
    const int gNw  = (int)(((size_t)Nn * 32 + T - 1) / T);
    const int gMM  = (Nn + MT - 1) / MT;

    // shared normalization (same for all 4 layers)
    deg_init <<<gN, T, 0, stream>>>(dinv, Nn);
    deg_accum<<<gE, T, 0, stream>>>(dinv, dst, ea, E);
    deg_fin  <<<gN, T, 0, stream>>>(dinv, Nn);
    norm_k   <<<gE, T, 0, stream>>>(nrm, dinv, src, dst, ea, E);

    // layer 1 (K = DIN)
    gemm_bf16_wmma<<<gMM, T, 0, stream>>>(x, W1, hbuf, Nn, DIN);
    init_out      <<<gNw, T, 0, stream>>>(act, hbuf, dinv, b1, Nn);
    edge_scatter  <<<gEw, T, 0, stream>>>(act, hbuf, nrm, src, dst, E);
    relu_k        <<<gNw, T, 0, stream>>>(act, Nn * 32);

    // layer 2
    gemm_bf16_wmma<<<gMM, T, 0, stream>>>(act, W2, hbuf, Nn, DH);
    init_out      <<<gNw, T, 0, stream>>>(act, hbuf, dinv, b2, Nn);
    edge_scatter  <<<gEw, T, 0, stream>>>(act, hbuf, nrm, src, dst, E);
    relu_k        <<<gNw, T, 0, stream>>>(act, Nn * 32);

    // layer 3
    gemm_bf16_wmma<<<gMM, T, 0, stream>>>(act, W3, hbuf, Nn, DH);
    init_out      <<<gNw, T, 0, stream>>>(act, hbuf, dinv, b3, Nn);
    edge_scatter  <<<gEw, T, 0, stream>>>(act, hbuf, nrm, src, dst, E);
    relu_k        <<<gNw, T, 0, stream>>>(act, Nn * 32);

    // layer 4 -> d_out, then log_softmax in place
    gemm_bf16_wmma<<<gMM, T, 0, stream>>>(act, W4, hbuf, Nn, DH);
    init_out      <<<gNw, T, 0, stream>>>(out, hbuf, dinv, b4, Nn);
    edge_scatter  <<<gEw, T, 0, stream>>>(out, hbuf, nrm, src, dst, E);
    log_softmax_k <<<gNw, T, 0, stream>>>(out, Nn);
}
